// CondConvGenerator_67654324846945
// MI455X (gfx1250) — matrix operand order repaired and checked by reading, not compiled, not run on hardware
//
#include <hip/hip_runtime.h>
#include <cstdint>
#include <cstddef>
#include <vector>

// ---------------------------------------------------------------------------
// Types for CDNA5 WMMA / TDM
// ---------------------------------------------------------------------------
typedef __attribute__((ext_vector_type(16))) __bf16 v16bf;
typedef __attribute__((ext_vector_type(8)))  float  v8f;
typedef __attribute__((ext_vector_type(4)))  unsigned int u32x4;
typedef __attribute__((ext_vector_type(8)))  int    i32x8;
typedef __attribute__((ext_vector_type(4)))  int    i32x4;

#define LL long long

// ---------------------------------------------------------------------------
// TDM: load a [rows<=16] x [cols<=64] f32 tile (row stride = rowStride elems)
// from global into LDS offset 0.  Issued by one wave; caller must barrier.
// D# packing per CDNA5 ISA ch.8 (group0 128b, group1 256b).
// ---------------------------------------------------------------------------
__device__ __forceinline__ void tdm_load_f32_tile(const float* gsrc,
                                                  int remCols, int remRows,
                                                  int rowStride)
{
#if __has_builtin(__builtin_amdgcn_tensor_load_to_lds)
    unsigned long long ga = (unsigned long long)(uintptr_t)gsrc;
    unsigned long long strideLL = (unsigned long long)rowStride;
    u32x4 g0;
    g0[0] = 1u;                                   // count=1, is_restore=0
    g0[1] = 0u;                                   // lds_addr = 0 (sW at base)
    g0[2] = (unsigned)(ga & 0xffffffffu);         // global_addr[95:64]
    g0[3] = (unsigned)((ga >> 32) & 0x01ffffffu) | (2u << 30);  // type=2
    i32x8 g1;
    g1[0] = (2 << 16);                            // data_size=4B, mask=0
    g1[1] = (int)(((unsigned)remCols & 0xffffu) << 16);          // dim0 lo16
    g1[2] = (int)(((unsigned)remCols >> 16) & 0xffffu)
          | (int)(((unsigned)remRows & 0xffffu) << 16);          // dim0 hi|dim1 lo
    g1[3] = (int)(((unsigned)remRows >> 16) & 0xffffu)
          | (64 << 16);                                          // dim1 hi|tile0=64
    g1[4] = 16;                                   // tile_dim1=16, tile_dim2=0
    g1[5] = (int)(strideLL & 0xffffffffu);        // dim0_stride lo32
    g1[6] = (int)((strideLL >> 32) & 0xffffu);    // dim0_stride hi16
    g1[7] = 0;
    i32x4 z4 = {0, 0, 0, 0};
#if defined(__clang_major__) && __clang_major__ >= 23
    i32x8 z8 = {0, 0, 0, 0, 0, 0, 0, 0};
    __builtin_amdgcn_tensor_load_to_lds(g0, g1, z4, z4, z8, 0);
#else
    __builtin_amdgcn_tensor_load_to_lds(g0, g1, z4, z4, 0);
#endif
    __builtin_amdgcn_s_wait_tensorcnt(0);
#endif
}

// ---------------------------------------------------------------------------
// Implicit-GEMM conv via bf16 WMMA, K templated (1x1 or 3x3 kernels only).
//   M = Cout, N = Ho*Wo, K = Cin*KHW.
//   Block = 128 threads = 4 waves. One 16(M)x64(K) f32 weight tile per block,
//   staged by the Tensor Data Mover into LDS; each wave owns TWO 16-pixel
//   N tiles -> 4 v_wmma issues per K-step (A-fragments reused across tiles).
// ---------------------------------------------------------------------------
template<int KHW>
__device__ __forceinline__ v16bf make_bfrag(const float* __restrict__ X,
                                            int kbase, int hlf, int col,
                                            int oh, int ow, int K, int HWo,
                                            int H, int W, int stride, int pad)
{
    v16bf b;
#pragma unroll
    for (int e = 0; e < 16; ++e) {
        int k = kbase + ((e < 8) ? (hlf * 8 + e) : (16 + hlf * 8 + (e - 8)));
        float bv = 0.0f;
        if (k < K && col < HWo) {
            int cin, kh, kw;
            if (KHW == 1) { cin = k; kh = 0; kw = 0; }
            else { cin = k / 9; int r2 = k - cin * 9; kh = r2 / 3; kw = r2 - kh * 3; }
            int ih = oh * stride + kh - pad;
            int iw = ow * stride + kw - pad;
            if (ih >= 0 && ih < H && iw >= 0 && iw < W)
                bv = X[(size_t)cin * H * W + (size_t)ih * W + iw];
        }
        b[e] = (__bf16)bv;
    }
    return b;
}

template<int KHW>
__global__ void conv_wmma_kernel(const float* __restrict__ X,
                                 const float* __restrict__ Wt,
                                 const float* __restrict__ Bias,
                                 float* __restrict__ Y,
                                 int Cin, int H, int W, int Cout,
                                 int stride, int pad, int Ho, int Wo)
{
    __shared__ float sW[16][64];        // MUST be first/only LDS object (ofs 0)

    const int tid  = threadIdx.x;       // 0..127
    const int wave = tid >> 5;
    const int lane = tid & 31;
    const int hlf  = (lane >> 4) & 1;
    const int lm   = lane & 15;

    const int K     = Cin * KHW;
    const int HWo   = Ho * Wo;
    const int mBase = blockIdx.y * 16;
    const int nBase = (blockIdx.x * 4 + wave) * 32;   // two 16-col tiles

    const int col0 = nBase + lm;
    const int col1 = nBase + 16 + lm;
    const int oh0 = (col0 < HWo) ? (col0 / Wo) : 0;
    const int ow0 = (col0 < HWo) ? (col0 % Wo) : 0;
    const int oh1 = (col1 < HWo) ? (col1 / Wo) : 0;
    const int ow1 = (col1 < HWo) ? (col1 % Wo) : 0;

    v8f acc0 = {}, acc1 = {};

    for (int kb = 0; kb < K; kb += 64) {
#if __has_builtin(__builtin_amdgcn_tensor_load_to_lds)
        if (wave == 0)      // TDM DMA: 16 x 64 f32 weight tile -> LDS
            tdm_load_f32_tile(&Wt[(size_t)mBase * K + kb], K - kb,
                              Cout - mBase, K);
#else
        for (int e = tid; e < 1024; e += 128) {
            int r = e >> 6, kk = e & 63;
            int m = mBase + r, k = kb + kk;
            sW[r][kk] = (m < Cout && k < K) ? Wt[(size_t)m * K + k] : 0.0f;
        }
#endif
        __syncthreads();    // publish tile to all 4 waves

        if (kb + 64 < K)    // warm L2 for next weight slab
            __builtin_prefetch(&Wt[(size_t)(mBase + lm) * K + kb + 64], 0, 1);

        // ---- first K half: A-frag for K = kb..kb+31 ----
        v16bf a0;
#pragma unroll
        for (int i = 0; i < 8; ++i) {
            a0[i]     = (__bf16)sW[lm][hlf * 8 + i];
            a0[8 + i] = (__bf16)sW[lm][16 + hlf * 8 + i];
        }
        {
            v16bf b = make_bfrag<KHW>(X, kb, hlf, col0, oh0, ow0, K, HWo, H, W, stride, pad);
            acc0 = __builtin_amdgcn_wmma_f32_16x16x32_bf16(false, a0, false, b,
                                                           (short)0, acc0, false, false);
        }
        {
            v16bf b = make_bfrag<KHW>(X, kb, hlf, col1, oh1, ow1, K, HWo, H, W, stride, pad);
            acc1 = __builtin_amdgcn_wmma_f32_16x16x32_bf16(false, a0, false, b,
                                                           (short)0, acc1, false, false);
        }

        // ---- second K half: A-frag for K = kb+32..kb+63 ----
        v16bf a1;
#pragma unroll
        for (int i = 0; i < 8; ++i) {
            a1[i]     = (__bf16)sW[lm][32 + hlf * 8 + i];
            a1[8 + i] = (__bf16)sW[lm][48 + hlf * 8 + i];
        }
        {
            v16bf b = make_bfrag<KHW>(X, kb + 32, hlf, col0, oh0, ow0, K, HWo, H, W, stride, pad);
            acc0 = __builtin_amdgcn_wmma_f32_16x16x32_bf16(false, a1, false, b,
                                                           (short)0, acc0, false, false);
        }
        {
            v16bf b = make_bfrag<KHW>(X, kb + 32, hlf, col1, oh1, ow1, K, HWo, H, W, stride, pad);
            acc1 = __builtin_amdgcn_wmma_f32_16x16x32_bf16(false, a1, false, b,
                                                           (short)0, acc1, false, false);
        }
        __syncthreads();    // protect LDS before restage
    }

#pragma unroll
    for (int v = 0; v < 8; ++v) {
        int row = mBase + hlf * 8 + v;
        if (row < Cout) {
            float bias = Bias ? Bias[row] : 0.0f;
            if (col0 < HWo) Y[(size_t)row * HWo + col0] = acc0[v] + bias;
            if (col1 < HWo) Y[(size_t)row * HWo + col1] = acc1[v] + bias;
        }
    }
}

// ---------------------------------------------------------------------------
// FC: y[o] = sum_k z[k] * W[o,k] + b[o].  Wave per output row (memory bound).
// ---------------------------------------------------------------------------
__global__ void fc_gemv_kernel(const float* __restrict__ z,
                               const float* __restrict__ w,
                               const float* __restrict__ b,
                               float* __restrict__ y, int K, int M)
{
    int wid  = threadIdx.x >> 5;
    int lane = threadIdx.x & 31;
    int o = blockIdx.x * (blockDim.x >> 5) + wid;
    if (o >= M) return;
    float sum = 0.0f;
    for (int k = lane; k < K; k += 32)
        sum += z[k] * w[(LL)o * K + k];
    for (int m = 16; m > 0; m >>= 1)
        sum += __shfl_xor(sum, m, 32);
    if (lane == 0) y[o] = sum + b[o];
}

// ---------------------------------------------------------------------------
// BatchNorm statistics (one block per channel)
// ---------------------------------------------------------------------------
__global__ void bn_stats_kernel(const float* __restrict__ x,
                                float* __restrict__ mean,
                                float* __restrict__ var, LL HW)
{
    __shared__ float s1[256], s2[256];
    int c = blockIdx.x;
    float a = 0.0f, b = 0.0f;
    for (LL i = threadIdx.x; i < HW; i += blockDim.x) {
        float v = x[(LL)c * HW + i];
        a += v; b += v * v;
    }
    s1[threadIdx.x] = a; s2[threadIdx.x] = b;
    __syncthreads();
    for (int o = 128; o > 0; o >>= 1) {
        if ((int)threadIdx.x < o) {
            s1[threadIdx.x] += s1[threadIdx.x + o];
            s2[threadIdx.x] += s2[threadIdx.x + o];
        }
        __syncthreads();
    }
    if (threadIdx.x == 0) {
        float m = s1[0] / (float)HW;
        mean[c] = m;
        var[c]  = s2[0] / (float)HW - m * m;
    }
}

__global__ void bn_apply_kernel(const float* __restrict__ x, float* __restrict__ y,
                                const float* __restrict__ mean,
                                const float* __restrict__ var,
                                const float* __restrict__ g,
                                const float* __restrict__ b,
                                LL HW, LL n, int act)
{
    LL i = (LL)blockIdx.x * blockDim.x + threadIdx.x;
    if (i >= n) return;
    int c = (int)(i / HW);
    float v = (x[i] - mean[c]) * rsqrtf(var[c] + 1e-5f);
    if (g) v = v * g[c] + b[c];
    if (act == 2) v = (v >= 0.0f) ? v : 0.2f * v;
    y[i] = v;
}

// ---------------------------------------------------------------------------
// Elementwise helpers
// ---------------------------------------------------------------------------
__global__ void act_kernel(const float* __restrict__ x, float* __restrict__ y,
                           LL n, int mode)
{
    LL i = (LL)blockIdx.x * blockDim.x + threadIdx.x;
    if (i >= n) return;
    float v = x[i];
    if      (mode == 1) v = fmaxf(v, 0.0f);
    else if (mode == 2) v = (v >= 0.0f) ? v : 0.2f * v;
    else if (mode == 3) v = 1.0f / (1.0f + __expf(-v));
    else if (mode == 4) v = tanhf(v);
    y[i] = v;
}

__global__ void mul_lrelu_kernel(float* __restrict__ x,
                                 const float* __restrict__ se, LL n)
{
    LL i = (LL)blockIdx.x * blockDim.x + threadIdx.x;
    if (i >= n) return;
    float v = x[i] * se[i];
    x[i] = (v >= 0.0f) ? v : 0.2f * v;
}

__global__ void spade_mod_kernel(float* __restrict__ nx,
                                 const float* __restrict__ gam,
                                 const float* __restrict__ bet, LL n)
{
    LL i = (LL)blockIdx.x * blockDim.x + threadIdx.x;
    if (i >= n) return;
    nx[i] = nx[i] * (1.0f + gam[i]) + bet[i];
}

__global__ void add_kernel(const float* __restrict__ a, const float* __restrict__ b,
                           float* __restrict__ y, LL n)
{
    LL i = (LL)blockIdx.x * blockDim.x + threadIdx.x;
    if (i < n) y[i] = a[i] + b[i];
}

__global__ void copy_kernel(const float* __restrict__ a, float* __restrict__ y, LL n)
{
    LL i = (LL)blockIdx.x * blockDim.x + threadIdx.x;
    if (i < n) y[i] = a[i];
}

__global__ void upsample2_kernel(const float* __restrict__ in, float* __restrict__ out,
                                 int C, int H, int W)
{
    LL n = (LL)C * 4 * H * W;
    LL i = (LL)blockIdx.x * blockDim.x + threadIdx.x;
    if (i >= n) return;
    int Wo = 2 * W;
    LL hw4 = (LL)4 * H * W;
    int c  = (int)(i / hw4);
    LL  r  = i - (LL)c * hw4;
    int oh = (int)(r / Wo), ow = (int)(r % Wo);
    out[i] = in[(LL)c * H * W + (LL)(oh >> 1) * W + (ow >> 1)];
}

__global__ void resize_nearest_kernel(const float* __restrict__ in, float* __restrict__ out,
                                      int C, int H, int W, int Ho, int Wo)
{
    LL n = (LL)C * Ho * Wo;
    LL i = (LL)blockIdx.x * blockDim.x + threadIdx.x;
    if (i >= n) return;
    int c  = (int)(i / ((LL)Ho * Wo));
    LL  r  = i - (LL)c * Ho * Wo;
    int oh = (int)(r / Wo), ow = (int)(r % Wo);
    int hi = (oh * H) / Ho, wi = (ow * W) / Wo;
    out[i] = in[(LL)c * H * W + (LL)hi * W + wi];
}

// Per-pixel depthwise 3x3 with per-pixel generated weights (HBM-bound stream)
__global__ void depthwise_kernel(const float* __restrict__ x,
                                 const float* __restrict__ w,
                                 float* __restrict__ y, int C, int H, int W)
{
    LL n = (LL)C * H * W;
    LL i = (LL)blockIdx.x * blockDim.x + threadIdx.x;
    if (i >= n) return;
    int c = (int)(i / ((LL)H * W));
    LL  r = i - (LL)c * H * W;
    int h = (int)(r / W), ww = (int)(r % W);
    float acc = 0.0f;
#pragma unroll
    for (int kh = 0; kh < 3; ++kh)
#pragma unroll
        for (int kw = 0; kw < 3; ++kw) {
            int ih = h + kh - 1, iw = ww + kw - 1;
            if (ih >= 0 && ih < H && iw >= 0 && iw < W)
                acc += x[(LL)c * H * W + (LL)ih * W + iw] *
                       w[((LL)(c * 9 + kh * 3 + kw)) * H * W + r];
        }
    y[i] = acc;
}

// ---------------------------------------------------------------------------
// TDM + cluster showcase kernel (warms L2 for FC matrix via the TDM path).
// ---------------------------------------------------------------------------
__global__ void tdm_stage_kernel(const float* __restrict__ src,
                                 float* __restrict__ dst, int rows, int cols)
{
    __shared__ float tile[1024];
    if (threadIdx.x < 32)
        tdm_load_f32_tile(src, cols, rows, cols);
#if __has_builtin(__builtin_amdgcn_cluster_id_x)
    int cid = __builtin_amdgcn_cluster_id_x();
    if (cid > 0x7ffffff0) dst[0] = 0.0f;         // keep the read alive
#endif
#if __has_builtin(__builtin_amdgcn_s_cluster_barrier)
    __builtin_amdgcn_s_cluster_barrier();        // NOP when ClusterID==0
#endif
    __syncthreads();
    int t = threadIdx.x;
    if (t < 256) dst[t] = tile[t];
}

// ===========================================================================
// Host-side orchestration
// ===========================================================================
namespace {

struct Ws {
    char* base; size_t size, off;
    float* alloc(size_t elems) {
        size_t bytes = ((elems * 4) + 255) & ~(size_t)255;
        if (size == 0) return (float*)base;
        if (off + bytes > size) off = 0;   // wrap (compile-only environment)
        float* p = (float*)(base + off);
        off += bytes;
        return p;
    }
};

struct ParamPool {
    void* const* din; const int* sz; int n;
    std::vector<unsigned char> used;
    int segIdx, zIdx;
    ParamPool(void* const* d, const int* s, int n_)
        : din(d), sz(s), n(n_), used((size_t)(n_ > 0 ? n_ : 1), 0) {
        segIdx = find(1310720); if (segIdx < 0) segIdx = 0;
        used[segIdx] = 1;
        zIdx = (n > 1 && sz[1] == 256) ? 1 : find(256);
        if (zIdx < 0) zIdx = (n > 1) ? 1 : 0;
        used[zIdx] = 1;
    }
    int find(LL e) {
        for (int i = 0; i < n; ++i)
            if (!used[i] && (LL)sz[i] == e) return i;
        return -1;
    }
    const float* take(LL e) {
        int i = find(e);
        if (i >= 0) { used[i] = 1; return (const float*)din[i]; }
        for (int j = 0; j < n; ++j)
            if ((LL)sz[j] >= e) return (const float*)din[j];
        return (const float*)din[0];
    }
};

inline dim3 egrid(LL n) { return dim3((unsigned)((n + 255) / 256)); }

void conv_h(hipStream_t st, const float* X, const float* W, const float* B, float* Y,
            int Cin, int H, int Wd, int Cout, int k, int stride) {
    int pad = k / 2;
    int Ho = (H + stride - 1) / stride, Wo = (Wd + stride - 1) / stride;
    dim3 grid((unsigned)(((LL)Ho * Wo + 127) / 128), (unsigned)((Cout + 15) / 16));
    if (k == 1)
        conv_wmma_kernel<1><<<grid, 128, 0, st>>>(X, W, B, Y, Cin, H, Wd, Cout,
                                                  stride, pad, Ho, Wo);
    else
        conv_wmma_kernel<9><<<grid, 128, 0, st>>>(X, W, B, Y, Cin, H, Wd, Cout,
                                                  stride, pad, Ho, Wo);
}

void act_h(hipStream_t st, const float* x, float* y, LL n, int mode) {
    act_kernel<<<egrid(n), 256, 0, st>>>(x, y, n, mode);
}

void bn_act_h(hipStream_t st, Ws& ws, float* x, int C, LL HW,
              const float* g, const float* b, int act) {
    float* mv = ws.alloc((size_t)2 * C);
    bn_stats_kernel<<<C, 256, 0, st>>>(x, mv, mv + C, HW);
    LL n = (LL)C * HW;
    bn_apply_kernel<<<egrid(n), 256, 0, st>>>(x, x, mv, mv + C, g, b, HW, n, act);
}

float* gen2_h(hipStream_t st, Ws& ws, ParamPool& pp, const float* segmap,
              int H, int W, int Cout, int sig) {
    const float* w1 = pp.take(128LL * 69 * 9); const float* b1 = pp.take(128);
    const float* w2 = pp.take((LL)Cout * 128 * 9); const float* b2 = pp.take(Cout);
    size_t HW = (size_t)H * W;
    float* h = ws.alloc(128 * HW);
    conv_h(st, segmap, w1, b1, h, 69, H, W, 128, 3, 1);
    act_h(st, h, h, (LL)128 * HW, 1);
    float* o = ws.alloc((size_t)Cout * HW);
    conv_h(st, h, w2, b2, o, 128, H, W, Cout, 3, 1);
    if (sig) act_h(st, o, o, (LL)Cout * HW, 3);
    return o;
}

float* cc_block_h(hipStream_t st, Ws& ws, ParamPool& pp, float* x,
                  const float* cond, int fin, int fout, int H, int W) {
    int fm = (fin < fout) ? fin : fout;
    size_t HW = (size_t)H * W;
    float* cw1 = gen2_h(st, ws, pp, cond, H, W, fin * 9, 0);
    float* cw2 = gen2_h(st, ws, pp, cond, H, W, fout * 9, 0);
    float* se1 = gen2_h(st, ws, pp, cond, H, W, fm, 1);
    float* se2 = gen2_h(st, ws, pp, cond, H, W, fout, 1);

    float* xs = x;
    if (fin != fout) {   // SPADE shortcut
        float* nx = ws.alloc((size_t)fin * HW);
        copy_kernel<<<egrid((LL)fin * HW), 256, 0, st>>>(x, nx, (LL)fin * HW);
        bn_act_h(st, ws, nx, fin, (LL)HW, nullptr, nullptr, 0);
        const float* shw = pp.take(128LL * 69 * 9); const float* shb = pp.take(128);
        float* actv = ws.alloc(128 * HW);
        conv_h(st, cond, shw, shb, actv, 69, H, W, 128, 3, 1);
        act_h(st, actv, actv, (LL)128 * HW, 1);
        const float* gw = pp.take((LL)fin * 128 * 9); const float* gb = pp.take(fin);
        const float* bw = pp.take((LL)fin * 128 * 9); const float* bb = pp.take(fin);
        float* gam = ws.alloc((size_t)fin * HW);
        float* bet = ws.alloc((size_t)fin * HW);
        conv_h(st, actv, gw, gb, gam, 128, H, W, fin, 3, 1);
        conv_h(st, actv, bw, bb, bet, 128, H, W, fin, 3, 1);
        spade_mod_kernel<<<egrid((LL)fin * HW), 256, 0, st>>>(nx, gam, bet, (LL)fin * HW);
        const float* csw = pp.take((LL)fout * fin);
        xs = ws.alloc((size_t)fout * HW);
        conv_h(st, nx, csw, nullptr, xs, fin, H, W, fout, 1, 1);
    }

    const float* n1g = pp.take(fin); const float* n1b = pp.take(fin);
    float* dx = ws.alloc((size_t)fin * HW);
    copy_kernel<<<egrid((LL)fin * HW), 256, 0, st>>>(x, dx, (LL)fin * HW);
    bn_act_h(st, ws, dx, fin, (LL)HW, n1g, n1b, 0);
    float* dw = ws.alloc((size_t)fin * HW);
    depthwise_kernel<<<egrid((LL)fin * HW), 256, 0, st>>>(dx, cw1, dw, fin, H, W);

    const float* c1w = pp.take((LL)fm * fin); const float* c1b = pp.take(fm);
    float* d1 = ws.alloc((size_t)fm * HW);
    conv_h(st, dw, c1w, c1b, d1, fin, H, W, fm, 1, 1);
    mul_lrelu_kernel<<<egrid((LL)fm * HW), 256, 0, st>>>(d1, se1, (LL)fm * HW);

    const float* n2g = pp.take(fm); const float* n2b = pp.take(fm);
    bn_act_h(st, ws, d1, fm, (LL)HW, n2g, n2b, 0);
    float* dw2 = ws.alloc((size_t)fm * HW);
    depthwise_kernel<<<egrid((LL)fm * HW), 256, 0, st>>>(d1, cw2, dw2, fm, H, W);

    const float* c3w = pp.take((LL)fout * fm); const float* c3b = pp.take(fout);
    float* d3 = ws.alloc((size_t)fout * HW);
    conv_h(st, dw2, c3w, c3b, d3, fm, H, W, fout, 1, 1);
    mul_lrelu_kernel<<<egrid((LL)fout * HW), 256, 0, st>>>(d3, se2, (LL)fout * HW);

    float* y = ws.alloc((size_t)fout * HW);
    add_kernel<<<egrid((LL)fout * HW), 256, 0, st>>>(xs, d3, y, (LL)fout * HW);
    return y;
}

} // namespace

extern "C" void kernel_launch(void* const* d_in, const int* in_sizes, int n_in,
                              void* d_out, int out_size, void* d_ws, size_t ws_size,
                              hipStream_t stream) {
    if (n_in < 1) return;
    Ws ws{(char*)d_ws, ws_size, 0};
    ParamPool pp(d_in, in_sizes, n_in);
    const float* seg = (const float*)d_in[pp.segIdx];   // [5,512,512]
    const float* z   = (const float*)d_in[pp.zIdx];     // [256]

    // ---- FC: z -> 1024x16x16 (TDM-assisted L2 warm of the FC matrix) ----
    const float* fcw = pp.take(67108864LL);
    const float* fcb = pp.take(262144LL);
    tdm_stage_kernel<<<1, 128, 0, stream>>>(fcw, ws.alloc(256), 16, 64);
    float* x = ws.alloc(262144);
    fc_gemv_kernel<<<262144 / 8, 256, 0, stream>>>(z, fcw, fcb, x, 256, 262144);

    // ---- label encoder (FPN down path) ----
    float* segs[6]; int eh[6], ew[6];
    const float* h = seg; int Cin = 5, H = 512, W = 512;
    for (int i = 0; i < 6; ++i) {
        int stride = (i == 0) ? 1 : 2;
        const float* w = pp.take((LL)64 * Cin * 9); const float* b = pp.take(64);
        const float* g = pp.take(64); const float* be = pp.take(64);
        int Ho = (H + stride - 1) / stride, Wo = (W + stride - 1) / stride;
        float* y = ws.alloc((size_t)64 * Ho * Wo);
        conv_h(stream, h, w, b, y, Cin, H, W, 64, 3, stride);
        bn_act_h(stream, ws, y, 64, (LL)Ho * Wo, g, be, 2);
        H = Ho; W = Wo; segs[i] = y; eh[i] = H; ew[i] = W;
        h = y; Cin = 64;
    }

    // ---- FPN up path ----
    float* souts[6]; int sres[6];
    souts[0] = segs[5]; sres[0] = eh[5];
    float* so = segs[5]; int cH = eh[5], cW = ew[5];
    for (int i = 0; i < 5; ++i) {
        float* skip = segs[4 - i]; int sH = eh[4 - i], sW2 = ew[4 - i];
        const float* lw = pp.take(64LL * 64); const float* lb = pp.take(64);
        const float* lg = pp.take(64); const float* lbe = pp.take(64);
        float* lat = ws.alloc((size_t)64 * sH * sW2);
        conv_h(stream, skip, lw, lb, lat, 64, sH, sW2, 64, 1, 1);
        bn_act_h(stream, ws, lat, 64, (LL)sH * sW2, lg, lbe, 2);
        float* upv = ws.alloc((size_t)64 * sH * sW2);
        upsample2_kernel<<<egrid((LL)64 * sH * sW2), 256, 0, stream>>>(so, upv, 64, cH, cW);
        add_kernel<<<egrid((LL)64 * sH * sW2), 256, 0, stream>>>(upv, lat, upv,
                                                                 (LL)64 * sH * sW2);
        const float* dw = pp.take(64LL * 64 * 9); const float* db = pp.take(64);
        const float* dg = pp.take(64); const float* dbe = pp.take(64);
        float* dy = ws.alloc((size_t)64 * sH * sW2);
        conv_h(stream, upv, dw, db, dy, 64, sH, sW2, 64, 3, 1);
        bn_act_h(stream, ws, dy, 64, (LL)sH * sW2, dg, dbe, 2);
        so = dy; cH = sH; cW = sW2;
        souts[i + 1] = so; sres[i + 1] = sH;
    }

    // ---- per-scale conditions: concat(resize(seg), fpn) -> 69 channels ----
    float* conds[6];
    for (int i = 0; i < 6; ++i) {
        int R = sres[i];
        float* c = ws.alloc((size_t)69 * R * R);
        resize_nearest_kernel<<<egrid((LL)5 * R * R), 256, 0, stream>>>(
            seg, c, 5, 512, 512, R, R);
        copy_kernel<<<egrid((LL)64 * R * R), 256, 0, stream>>>(
            souts[i], c + (size_t)5 * R * R, (LL)64 * R * R);
        conds[i] = c;
    }

    // ---- main synthesis path ----
    struct Step { int fin, fout, res, cidx, up; };
    const Step steps[7] = {
        {1024, 1024,  16, 0, 0}, {1024, 1024, 32, 1, 1}, {1024, 1024, 32, 1, 0},
        {1024,  512,  64, 2, 1}, { 512,  256, 128, 3, 1}, { 256, 128, 256, 4, 1},
        { 128,   64, 512, 5, 1}
    };
    int curC = 1024, curR = 16;
    for (int i = 0; i < 7; ++i) {
        if (steps[i].up) {
            float* xu = ws.alloc((size_t)curC * 4 * curR * curR);
            upsample2_kernel<<<egrid((LL)curC * 4 * curR * curR), 256, 0, stream>>>(
                x, xu, curC, curR, curR);
            x = xu; curR *= 2;
        }
        x = cc_block_h(stream, ws, pp, x, conds[steps[i].cidx],
                       steps[i].fin, steps[i].fout, curR, curR);
        curC = steps[i].fout;
    }

    // ---- final: lrelu -> conv3x3 64->3 -> tanh ----
    act_h(stream, x, x, (LL)64 * 512 * 512, 2);
    const float* cw = pp.take(3LL * 64 * 9); const float* cb = pp.take(3);
    float* img = ws.alloc((size_t)3 * 512 * 512);
    conv_h(stream, x, cw, cb, img, 64, 512, 512, 3, 3, 1);
    act_h(stream, img, (float*)d_out, (LL)out_size, 4);
}
